// OthelloGPT_24172075942193
// MI455X (gfx1250) — matrix operand order repaired and checked by reading, compiled, not run on hardware
//
#include <hip/hip_runtime.h>

// ---------------- problem constants ----------------
#define D_MODEL 768
#define N_HEADS 12
#define D_HEAD  64
#define SEQ     512
#define BATCH   8
#define MROWS   (BATCH*SEQ)   // 4096
#define FFN     3072
#define VOCAB   66
#define NLAYER  8
#define GEMM_BM 128           // 8 waves x 16 rows per block

typedef __bf16 bf16;
typedef __attribute__((ext_vector_type(16))) __bf16 v16bf;
typedef __attribute__((ext_vector_type(8)))  float  v8f;

// D = A(16x32 bf16) x B(32x16 bf16) + C(16x16 f32), wave32
__device__ __forceinline__ v8f wmma_bf16(v16bf a, v16bf b, v8f c) {
  return __builtin_amdgcn_wmma_f32_16x16x32_bf16(
      /*neg_a=*/false, a, /*neg_b=*/false, b,
      /*c_mod=*/(short)0, c, /*reuse_a=*/false, /*reuse_b=*/false);
}

// fp32 -> bf16 hi/lo split: x ~= hi + lo, captures ~16 mantissa bits
__device__ __forceinline__ void split_bf16(float x, bf16& hi, bf16& lo) {
  bf16 h = (bf16)x;
  hi = h;
  lo = (bf16)(x - (float)h);
}

// A-fragment K-offset for 16-bit A 16x32 (contiguous order, matches B):
// lanes 0-15 hold K 0..15, lanes 16-31 hold K 16..31
__device__ __forceinline__ int a_koff(int i, int khalf) {
  return khalf * 16 + i;
}

// ---------------- embedding ----------------
__global__ __launch_bounds__(256) void embed_kernel(
    const float* __restrict__ tok, const float* __restrict__ pos,
    const int* __restrict__ ids, float* __restrict__ x)
{
  long i = (long)blockIdx.x * 256 + threadIdx.x;
  if (i >= (long)MROWS * D_MODEL) return;
  long row = i / D_MODEL;
  int  d   = (int)(i - row * D_MODEL);
  int  t   = (int)(row % SEQ);
  int  id  = ids[row];
  x[i] = tok[(long)id * D_MODEL + d] + pos[(long)t * D_MODEL + d];
}

// ---------------- layernorm (torch std_mean: ddof=1, eps on std) -------------
__global__ __launch_bounds__(256) void ln_kernel(
    const float* __restrict__ in, const float* __restrict__ g,
    const float* __restrict__ b, float* __restrict__ out, int Mdim)
{
  int row = blockIdx.x * 8 + (threadIdx.x >> 5);
  if (row >= Mdim) return;
  int lane = threadIdx.x & 31;
  const float* p = in + (long)row * D_MODEL;
  float sum = 0.0f;
  for (int i = lane; i < D_MODEL; i += 32) sum += p[i];
  for (int off = 16; off > 0; off >>= 1) sum += __shfl_xor(sum, off, 32);
  float mean = sum * (1.0f / D_MODEL);
  float vs = 0.0f;
  for (int i = lane; i < D_MODEL; i += 32) { float d = p[i] - mean; vs += d * d; }
  for (int off = 16; off > 0; off >>= 1) vs += __shfl_xor(vs, off, 32);
  float rs = 1.0f / (sqrtf(vs * (1.0f / (D_MODEL - 1))) + 1e-10f);
  float* q = out + (long)row * D_MODEL;
  for (int i = lane; i < D_MODEL; i += 32)
    q[i] = (p[i] - mean) * rs * g[i] + b[i];
}

// ---------------- block-tiled WMMA GEMM, double-buffered LDS staging ---------
// out[M,N] = act( A[M,K] x W + bias ) + resid
// W element for (k, col): W[(col/group)*K*group + (col%group) + k*group]
//   group == N  -> standard row-major [K,N]
//   group == 64 -> per-head [H, K, 64] QKV layout (col = h*64 + e)
// A block owns a 128x64 output tile (8 waves stacked in M, shared n0).
// Each 32(K)x64(N) W slab is loaded contiguously (b128), split to bf16 hi/lo
// once, and stored into LDS pre-swizzled into wave32 B-fragment order.
// Ping-pong buffering: next slab's global loads are issued before the current
// slab's WMMAs so load latency hides under the matrix pipe; 1 barrier/iter.
__global__ __launch_bounds__(256) void gemm_wmma(
    const float* __restrict__ A, const float* __restrict__ W,
    const float* __restrict__ bias, const float* __restrict__ resid,
    float* __restrict__ out, int Mdim, int Ndim, int Kdim, int group, int act)
{
  __shared__ v16bf bfrag_hi[2][4][32];   // [buf][n-subtile][lane] (8KB)
  __shared__ v16bf bfrag_lo[2][4][32];   // (8KB)

  const int tid   = threadIdx.x;
  const int lane  = tid & 31;
  const int wave  = tid >> 5;
  const int khalf = lane >> 4;
  const int l16   = lane & 15;

  const int ntiles_n = (Ndim + 63) >> 6;
  const int bn = blockIdx.x % ntiles_n;
  const int bm = blockIdx.x / ntiles_n;
  const int n0 = bn << 6;
  const int m0 = bm * GEMM_BM + wave * 16;

  // 64-wide N tile never crosses a `group` boundary (group==64 or group==N)
  const long wblk = (long)(n0 / group) * ((long)Kdim * group) + (n0 % group);
  const int  nrem = Ndim - n0;            // valid cols in this block
  const bool full = (nrem >= 64);

  // staging role: 32 rows x 64 cols = 2048 elems / 256 threads = 8 each
  const int kr  = tid >> 3;               // K row within slab (0..31)
  const int c0s = (tid & 7) * 8;          // first col handled (0..56)
  const int tln_half = 16 * (kr >> 4);    // lane-half from K row
  const int el = kr & 15;                 // fragment element index

  const float* arow = A + (long)(m0 + l16) * Kdim;

  // W slab loader (k0 = slab base); unpredicated loads, cndmask for tail tile
  auto load_w = [&](float (&xs)[8], int k0) {
    const float* wrow = W + wblk + (long)(k0 + kr) * group;
    if (full) {
      *(float4*)&xs[0] = *(const float4*)(wrow + c0s);
      *(float4*)&xs[4] = *(const float4*)(wrow + c0s + 4);
    } else {
      #pragma unroll
      for (int cc = 0; cc < 8; ++cc) {
        int c = c0s + cc;
        int csafe = (c < nrem) ? c : 0;
        float x = wrow[csafe];
        xs[cc] = (c < nrem) ? x : 0.0f;
      }
    }
  };
  // swizzle W slab into fragment order in LDS buffer `buf`
  auto store_w = [&](int buf, const float (&xs)[8]) {
    #pragma unroll
    for (int cc = 0; cc < 8; ++cc) {
      int c = c0s + cc;
      bf16 h, l; split_bf16(xs[cc], h, l);
      int t  = c >> 4;
      int ln = (c & 15) + tln_half;
      ((bf16*)&bfrag_hi[buf][t][ln])[el] = h;
      ((bf16*)&bfrag_lo[buf][t][ln])[el] = l;
    }
  };
  // A fragment: 1 contiguous 64B run per lane
  auto load_a = [&](float (&xs)[16], int k0) {
    const float* pa = arow + k0 + khalf * 16;
    *(float4*)&xs[0]  = *(const float4*)(pa);
    *(float4*)&xs[4]  = *(const float4*)(pa + 4);
    *(float4*)&xs[8]  = *(const float4*)(pa + 8);
    *(float4*)&xs[12] = *(const float4*)(pa + 12);
  };

  v8f acc[4] = {};
  float axs[16], wxs[8];

  // prologue: stage slab 0, preload A fragment 0
  load_w(wxs, 0);
  store_w(0, wxs);
  load_a(axs, 0);
  __syncthreads();

  for (int k0 = 0; k0 < Kdim; k0 += 32) {
    const int  cur  = (k0 >> 5) & 1;
    const int  nxt  = cur ^ 1;
    const bool more = (k0 + 32 < Kdim);

    // issue next-slab loads early: latency overlaps the WMMAs below
    float wn[8], an[16];
    if (more) {
      load_w(wn, k0 + 32);
      load_a(an, k0 + 32);
    }

    // current A fragment -> bf16 hi/lo
    v16bf ahi, alo;
    #pragma unroll
    for (int i = 0; i < 16; ++i) {
      // map pipeline layout to this code's operand order (contiguous k = 16*khalf + i)
      int src = (i < 8) ? i : (8 + (i - 8));      // identity: axs laid out as a_koff
      bf16 h, l; split_bf16(axs[src], h, l);
      ahi[i] = h; alo[i] = l;
    }

    #pragma unroll
    for (int t = 0; t < 4; ++t) {
      v16bf bhi = bfrag_hi[cur][t][lane];
      v16bf blo = bfrag_lo[cur][t][lane];
      // split-precision product: hi*hi + hi*lo + lo*hi (fp32 accumulate)
      acc[t] = wmma_bf16(alo, bhi, acc[t]);
      acc[t] = wmma_bf16(ahi, blo, acc[t]);
      acc[t] = wmma_bf16(ahi, bhi, acc[t]);
    }

    if (more) {
      store_w(nxt, wn);
      #pragma unroll
      for (int i = 0; i < 16; ++i) axs[i] = an[i];
    }
    __syncthreads();
  }

  if (m0 >= Mdim) return;
  #pragma unroll
  for (int t = 0; t < 4; ++t) {
    int cloc = t * 16 + l16;
    if (cloc >= nrem) continue;
    int coln = n0 + cloc;
    #pragma unroll
    for (int j = 0; j < 8; ++j) {
      int r = m0 + j + 8 * khalf;            // C/D layout: M = vgpr + 8*(lane>=16)
      float v = acc[t][j];
      if (bias) v += bias[coln];
      if (act == 1)                           // exact GELU
        v = 0.5f * v * (1.0f + erff(v * 0.70710678118654752440f));
      long oi = (long)r * Ndim + coln;
      if (resid) v += resid[oi];
      out[oi] = v;
    }
  }
}

// ---------------- fused causal attention ----------------
// One block (4 waves) per (batch, head, 16-row query tile).
// Scores for the whole row block live in 32KB LDS; softmax is done in place.
__global__ __launch_bounds__(128) void attn_fused(
    const float* __restrict__ q, const float* __restrict__ k,
    const float* __restrict__ v, float* __restrict__ o)
{
  __shared__ float s[16][SEQ];
  __shared__ float red[16][8];
  __shared__ float rowstat[16];

  const int tid   = threadIdx.x;
  const int lane  = tid & 31;
  const int wave  = tid >> 5;
  const int khalf = lane >> 4;
  const int l16   = lane & 15;

  const int qt   = blockIdx.x & 31;       // query tile (SEQ/16 = 32)
  const int bh   = blockIdx.x >> 5;
  const int hIdx = bh % N_HEADS;
  const int bIdx = bh / N_HEADS;

  const float scale = 0.036084391824351615f;  // 1/sqrt(768) (d_model!)

  const float* qbase = q + (long)bIdx * SEQ * D_MODEL + hIdx * D_HEAD;
  const float* kbase = k + (long)bIdx * SEQ * D_MODEL + hIdx * D_HEAD;
  const float* vbase = v + (long)bIdx * SEQ * D_MODEL + hIdx * D_HEAD;

  // ---- phase 1: S = (Q Kt) * scale, causal mask; waves stride over key tiles
  const float* ap = qbase + (long)(qt * 16 + l16) * D_MODEL;
  for (int kt = wave; kt < SEQ / 16; kt += 4) {
    if (kt > qt) {  // fully masked tile: zero-fill so phase 3 adds nothing
      for (int r = khalf; r < 16; r += 2) s[r][kt * 16 + l16] = 0.0f;
      continue;
    }
    const float* bp = kbase + (long)(kt * 16 + l16) * D_MODEL;
    v8f acc = {};
    #pragma unroll
    for (int k0 = 0; k0 < D_HEAD; k0 += 32) {
      v16bf ahi, alo, bhi, blo;
      #pragma unroll
      for (int i = 0; i < 16; ++i) {
        float xa = ap[k0 + a_koff(i, khalf)];
        bf16 h1, l1; split_bf16(xa, h1, l1);
        ahi[i] = h1; alo[i] = l1;
        float xb = bp[k0 + khalf * 16 + i];   // B = K^T: contiguous in head dim
        bf16 h2, l2; split_bf16(xb, h2, l2);
        bhi[i] = h2; blo[i] = l2;
      }
      acc = wmma_bf16(alo, bhi, acc);
      acc = wmma_bf16(ahi, blo, acc);
      acc = wmma_bf16(ahi, bhi, acc);
    }
    #pragma unroll
    for (int j = 0; j < 8; ++j) {
      int r    = j + 8 * khalf;
      int qrow = qt * 16 + r;
      int kc   = kt * 16 + l16;
      s[r][kc] = (kc <= qrow) ? acc[j] * scale : -__builtin_inff();
    }
  }
  __syncthreads();

  // ---- phase 2: in-place softmax over first kcount columns (fixed order)
  const int kcount = (qt + 1) * 16;
  const int r   = tid >> 3;
  const int seg = tid & 7;
  const int c0  = seg * 64;
  const int c1  = (c0 + 64 < kcount) ? (c0 + 64) : kcount;
  float mx = -__builtin_inff();
  for (int c = c0; c < c1; ++c) mx = fmaxf(mx, s[r][c]);
  red[r][seg] = mx;
  __syncthreads();
  if (tid < 16) {
    float mm = -__builtin_inff();
    for (int ss = 0; ss < 8; ++ss) mm = fmaxf(mm, red[tid][ss]);
    rowstat[tid] = mm;
  }
  __syncthreads();
  const float rm = rowstat[r];
  float sum = 0.0f;
  for (int c = c0; c < c1; ++c) { float e = expf(s[r][c] - rm); s[r][c] = e; sum += e; }
  red[r][seg] = sum;
  __syncthreads();
  if (tid < 16) {
    float ssum = 0.0f;
    for (int ss = 0; ss < 8; ++ss) ssum += red[tid][ss];
    rowstat[tid] = 1.0f / ssum;
  }
  __syncthreads();
  const float inv = rowstat[r];
  for (int c = c0; c < c1; ++c) s[r][c] *= inv;
  __syncthreads();

  // ---- phase 3: O = P(16xSEQ) x V(SEQx64); each wave owns 16 output cols
  {
    const int ecol = wave * 16 + l16;
    const float* vp = vbase + ecol;
    v8f acc = {};
    for (int k0 = 0; k0 < SEQ; k0 += 32) {
      v16bf ahi, alo, bhi, blo;
      #pragma unroll
      for (int i = 0; i < 16; ++i) {
        float xa = s[l16][k0 + a_koff(i, khalf)];
        bf16 h1, l1; split_bf16(xa, h1, l1);
        ahi[i] = h1; alo[i] = l1;
        int tk = k0 + khalf * 16 + i;
        float xb = vp[(long)tk * D_MODEL];
        bf16 h2, l2; split_bf16(xb, h2, l2);
        bhi[i] = h2; blo[i] = l2;
      }
      acc = wmma_bf16(alo, bhi, acc);
      acc = wmma_bf16(ahi, blo, acc);
      acc = wmma_bf16(ahi, bhi, acc);
    }
    #pragma unroll
    for (int j = 0; j < 8; ++j) {
      int qrow = qt * 16 + j + 8 * khalf;
      o[((long)(bIdx * SEQ + qrow)) * D_MODEL + hIdx * D_HEAD + ecol] = acc[j];
    }
  }
}

// ---------------- loss ----------------
__global__ __launch_bounds__(256) void nll_kernel(
    const float* __restrict__ logits, const int* __restrict__ tgt,
    float* __restrict__ nll)
{
  int row = blockIdx.x * 8 + (threadIdx.x >> 5);
  if (row >= MROWS) return;
  int lane = threadIdx.x & 31;
  const float* p = logits + (long)row * VOCAB;
  float m = -__builtin_inff();
  for (int i = lane; i < VOCAB; i += 32) m = fmaxf(m, p[i]);
  for (int off = 16; off > 0; off >>= 1) m = fmaxf(m, __shfl_xor(m, off, 32));
  float sum = 0.0f;
  for (int i = lane; i < VOCAB; i += 32) sum += expf(p[i] - m);
  for (int off = 16; off > 0; off >>= 1) sum += __shfl_xor(sum, off, 32);
  if (lane == 0) {
    int t = tgt[row];
    nll[row] = -(p[t] - m - logf(sum));
  }
}

__global__ __launch_bounds__(256) void reduce_kernel(
    const float* __restrict__ nll, float* __restrict__ loss)
{
  __shared__ float sh[256];
  float s = 0.0f;
  for (int i = threadIdx.x; i < MROWS; i += 256) s += nll[i];
  sh[threadIdx.x] = s;
  __syncthreads();
  for (int off = 128; off > 0; off >>= 1) {
    if ((int)threadIdx.x < off) sh[threadIdx.x] += sh[threadIdx.x + off];
    __syncthreads();
  }
  if (threadIdx.x == 0) loss[0] = sh[0] / (float)MROWS;
}

// ---------------- host orchestration ----------------
static inline void gemm_go(const float* A, const float* W, const float* bias,
                           const float* resid, float* out, int M, int N, int K,
                           int group, int act, hipStream_t stream)
{
  int ntiles_n = (N + 63) / 64;
  int ntiles_m = (M + GEMM_BM - 1) / GEMM_BM;
  gemm_wmma<<<ntiles_n * ntiles_m, 256, 0, stream>>>(A, W, bias, resid, out, M, N, K, group, act);
}

extern "C" void kernel_launch(void* const* d_in, const int* in_sizes, int n_in,
                              void* d_out, int out_size, void* d_ws, size_t ws_size,
                              hipStream_t stream)
{
  (void)in_sizes; (void)n_in; (void)out_size; (void)ws_size;

  const float* tok   = (const float*)d_in[0];
  const float* pos   = (const float*)d_in[1];
  const float* lnf_g = (const float*)d_in[2];
  const float* lnf_b = (const float*)d_in[3];
  const float* Wu    = (const float*)d_in[4];
  const float* bu    = (const float*)d_in[5];
  const int*   ids   = (const int*)d_in[6 + NLAYER * 13];
  const int*   tgt   = (const int*)d_in[7 + NLAYER * 13];

  float* logits = (float*)d_out;                       // [MROWS, VOCAB]
  float* loss   = logits + (long)MROWS * VOCAB;        // scalar

  char* ws = (char*)d_ws;
  const size_t MD = (size_t)MROWS * D_MODEL * sizeof(float);
  float* x   = (float*)(ws + 0 * MD);
  float* h   = (float*)(ws + 1 * MD);
  float* qb  = (float*)(ws + 2 * MD);
  float* kb  = (float*)(ws + 3 * MD);
  float* vb  = (float*)(ws + 4 * MD);
  float* ob  = (float*)(ws + 5 * MD);
  float* hid = (float*)(ws + 6 * MD);
  float* nll = (float*)(ws + 6 * MD + (size_t)MROWS * FFN * sizeof(float));

  embed_kernel<<<((long)MROWS * D_MODEL + 255) / 256, 256, 0, stream>>>(tok, pos, ids, x);

  for (int l = 0; l < NLAYER; ++l) {
    const int base = 6 + l * 13;
    const float* ln1_g = (const float*)d_in[base + 0];
    const float* ln1_b = (const float*)d_in[base + 1];
    const float* Wq    = (const float*)d_in[base + 2];
    const float* Wk    = (const float*)d_in[base + 3];
    const float* Wv    = (const float*)d_in[base + 4];
    const float* Wo    = (const float*)d_in[base + 5];
    const float* bo    = (const float*)d_in[base + 6];
    const float* ln2_g = (const float*)d_in[base + 7];
    const float* ln2_b = (const float*)d_in[base + 8];
    const float* W1    = (const float*)d_in[base + 9];
    const float* b1    = (const float*)d_in[base + 10];
    const float* W2    = (const float*)d_in[base + 11];
    const float* b2    = (const float*)d_in[base + 12];

    ln_kernel<<<MROWS / 8, 256, 0, stream>>>(x, ln1_g, ln1_b, h, MROWS);
    gemm_go(h, Wq, nullptr, nullptr, qb, MROWS, D_MODEL, D_MODEL, D_HEAD, 0, stream);
    gemm_go(h, Wk, nullptr, nullptr, kb, MROWS, D_MODEL, D_MODEL, D_HEAD, 0, stream);
    gemm_go(h, Wv, nullptr, nullptr, vb, MROWS, D_MODEL, D_MODEL, D_HEAD, 0, stream);
    attn_fused<<<BATCH * N_HEADS * (SEQ / 16), 128, 0, stream>>>(qb, kb, vb, ob);
    gemm_go(ob, Wo, bo, x, x, MROWS, D_MODEL, D_MODEL, D_MODEL, 0, stream);
    ln_kernel<<<MROWS / 8, 256, 0, stream>>>(x, ln2_g, ln2_b, h, MROWS);
    gemm_go(h, W1, b1, nullptr, hid, MROWS, FFN, D_MODEL, FFN, 1, stream);
    gemm_go(hid, W2, b2, x, x, MROWS, D_MODEL, FFN, D_MODEL, 0, stream);
  }

  ln_kernel<<<MROWS / 8, 256, 0, stream>>>(x, lnf_g, lnf_b, h, MROWS);
  gemm_go(h, Wu, bu, nullptr, logits, MROWS, VOCAB, D_MODEL, VOCAB, 0, stream);
  nll_kernel<<<MROWS / 8, 256, 0, stream>>>(logits, tgt, nll);
  reduce_kernel<<<1, 256, 0, stream>>>(nll, loss);
}
